// seq2seq_21827023798501
// MI455X (gfx1250) — matrix operand order repaired and checked by reading, not compiled, not run on hardware
//
#include <hip/hip_runtime.h>

typedef _Float16 f16;
typedef __attribute__((ext_vector_type(16))) _Float16 v16h;
typedef __attribute__((ext_vector_type(8)))  _Float16 v8h;
typedef __attribute__((ext_vector_type(8)))  float    v8f;

#define GRU_H  30
#define GRU_HP 32

// ---- activations: prefer hardware v_tanh_f32 (CDNA5 TRANS op), fall back to exp/rcp ----
#if defined(__has_builtin)
#if __has_builtin(__builtin_amdgcn_tanhf)
#define USE_HW_TANH 1
#endif
#endif

#ifdef USE_HW_TANH
__device__ __forceinline__ float ftanh_(float x) { return __builtin_amdgcn_tanhf(x); }
__device__ __forceinline__ float fsig(float x)   { return __builtin_fmaf(0.5f, __builtin_amdgcn_tanhf(0.5f * x), 0.5f); }
#else
__device__ __forceinline__ float fsig(float x) {
  float e = __expf(-x);
  return __builtin_amdgcn_rcpf(1.0f + e);
}
__device__ __forceinline__ float ftanh_(float x) {
  float e = __expf(-2.0f * x);                 // tanh(x) = 1 - 2e/(1+e), e=exp(-2x)
  return 1.0f - 2.0f * __builtin_amdgcn_rcpf(1.0f + e);
}
#endif

// Build one 32(K) x 16(N) f16 B-tile for v_wmma_f32_16x16x32_f16.
// B[k][n] = Wpad[16*t + n][k], Wpad = (96x32) zero-pad of row-major (90x30) W.
// B VGPR layout (wave32), matching A: lanes 0-15 hold N=lane, K=0..7,16..23;
// lanes 16-31 hold N=lane-16, K=8..15,24..31.  => element i of v16h = K = 8*(lane>>4) + (i&7) + 16*(i>>3).
__device__ __forceinline__ v16h build_btile(const float* W, int t, int lane) {
  const int n   = lane & 15;
  const int khi = (lane >> 4) << 3;
  const int j   = 16 * t + n;        // padded row 0..95
  const int g   = j >> 5;            // gate 0..2 (r,z,n)
  const int u   = j & 31;            // padded unit 0..31
  v16h b;
#pragma unroll
  for (int i = 0; i < 16; ++i) {
    const int k = khi + (i & 7) + ((i >> 3) << 4);
    float w = 0.0f;
    if (u < GRU_H && k < GRU_H) w = W[(g * GRU_H + u) * GRU_H + k];
    b[i] = (f16)w;
  }
  return b;
}

// A-matrix (16x32 f16) with every row equal to the staged 32-wide h vector.
// 16-bit A layout: lanes 0-15 get K=0..7 (v0-3) and K=16..23 (v4-7);
// lanes 16-31 get K=8..15 and K=24..31.  Two ds_load_b128 per lane.
__device__ __forceinline__ v16h load_arow(const f16* buf, int lane) {
  const int ho = (lane >> 4) << 3;              // 0 or 8 halves (0 or 16 bytes)
  v8h lo = *(const v8h*)(buf + ho);
  v8h hi = *(const v8h*)(buf + 16 + ho);
  v16h a;
#pragma unroll
  for (int i = 0; i < 8; ++i) { a[i] = lo[i]; a[8 + i] = hi[i]; }
  return a;
}

#define WMMA(A_, B_) \
  __builtin_amdgcn_wmma_f32_16x16x32_f16(false, (A_), false, (B_), (short)0, cz, false, false)

__global__ __launch_bounds__(32) void seq2seq_gru_kernel(
    const float* __restrict__ x, int n_enc,
    const float* __restrict__ eWih, const float* __restrict__ eWhh,
    const float* __restrict__ ebih, const float* __restrict__ ebhh,
    const float* __restrict__ dWih, const float* __restrict__ dWhh,
    const float* __restrict__ dbih, const float* __restrict__ dbhh,
    const float* __restrict__ linW, const float* __restrict__ linb,
    float* __restrict__ out, int n_dec)
{
  const int lane = threadIdx.x;
  const int a  = lane & 15;     // first hidden unit this lane owns
  const int ub = a + 16;        // second hidden unit this lane owns
  __shared__ __align__(64) f16 hX[GRU_HP];
  __shared__ __align__(64) f16 hH[GRU_HP];
  const v8f cz = {};

  // ---------------- encoder ----------------
  v16h Be[6];
#pragma unroll
  for (int t = 0; t < 6; ++t) Be[t] = build_btile(eWhh, t, lane);

  float ewih[3][2], ebi[3][2], ebh[3][2];
#pragma unroll
  for (int g = 0; g < 3; ++g)
#pragma unroll
    for (int s = 0; s < 2; ++s) {
      const int u = a + 16 * s;
      const bool v = (u < GRU_H);
      const int idx = g * GRU_H + u;
      ewih[g][s] = v ? eWih[idx] : 0.0f;   // enc_Wih is (90,1)
      ebi[g][s]  = v ? ebih[idx] : 0.0f;
      ebh[g][s]  = v ? ebhh[idx] : 0.0f;
    }

  float hA = 0.0f, hB = 0.0f;              // padded units start (and stay) 0
  for (int c0 = 0; c0 < n_enc; c0 += 32) {
    const float xv = (c0 + lane < n_enc) ? x[c0 + lane] : 0.0f;
    const int jmax = (n_enc - c0 < 32) ? (n_enc - c0) : 32;
    for (int j = 0; j < jmax; ++j) {
      const float xt = __shfl(xv, j, 32);  // uniform broadcast of x[t]
      hH[a]  = (f16)hA;                    // lanes 16-31 duplicate same value
      hH[ub] = (f16)hB;
      asm volatile("s_wait_dscnt 0x0" ::: "memory");
      const v16h A = load_arow(hH, lane);
      v8f c0r = WMMA(A, Be[0]); v8f c1r = WMMA(A, Be[1]);
      v8f c2r = WMMA(A, Be[2]); v8f c3r = WMMA(A, Be[3]);
      v8f c4r = WMMA(A, Be[4]); v8f c5r = WMMA(A, Be[5]);
      // unit a: tiles 0/2/4 ; unit b: tiles 1/3/5  (value at D row0 = VGPR0)
      const float rA = fsig (xt * ewih[0][0] + ebi[0][0] + c0r[0] + ebh[0][0]);
      const float zA = fsig (xt * ewih[1][0] + ebi[1][0] + c2r[0] + ebh[1][0]);
      const float nA = ftanh_(xt * ewih[2][0] + ebi[2][0] + rA * (c4r[0] + ebh[2][0]));
      hA = (1.0f - zA) * nA + zA * hA;
      const float rB = fsig (xt * ewih[0][1] + ebi[0][1] + c1r[0] + ebh[0][1]);
      const float zB = fsig (xt * ewih[1][1] + ebi[1][1] + c3r[0] + ebh[1][1]);
      const float nB = ftanh_(xt * ewih[2][1] + ebi[2][1] + rB * (c5r[0] + ebh[2][1]));
      hB = (1.0f - zB) * nB + zB * hB;
    }
  }

  // ---------------- decoder ----------------
  v16h Bw[6], Bh[6];
#pragma unroll
  for (int t = 0; t < 6; ++t) { Bw[t] = build_btile(dWih, t, lane);
                                Bh[t] = build_btile(dWhh, t, lane); }
  float dbi[3][2], dbh[3][2];
#pragma unroll
  for (int g = 0; g < 3; ++g)
#pragma unroll
    for (int s = 0; s < 2; ++s) {
      const int u = a + 16 * s;
      const bool v = (u < GRU_H);
      const int idx = g * GRU_H + u;
      dbi[g][s] = v ? dbih[idx] : 0.0f;
      dbh[g][s] = v ? dbhh[idx] : 0.0f;
    }
  const float lwA = linW[a];                        // a <= 15 < 30 always valid
  const float lwB = (ub < GRU_H) ? linW[ub] : 0.0f;
  const float lb  = linb[0];

  float hxA = hA, hxB = hB;                         // hx0 = h_enc
  float hhA = 1.0f, hhB = (ub < GRU_H) ? 1.0f : 0.0f;  // hh0 = ones (padded 0)

  for (int s = 0; s < n_dec; ++s) {
    hX[a] = (f16)hxA; hX[ub] = (f16)hxB;
    hH[a] = (f16)hhA; hH[ub] = (f16)hhB;
    asm volatile("s_wait_dscnt 0x0" ::: "memory");
    const v16h Ax = load_arow(hX, lane);
    const v16h Ah = load_arow(hH, lane);
    v8f i0 = WMMA(Ax, Bw[0]); v8f i1 = WMMA(Ax, Bw[1]);
    v8f i2 = WMMA(Ax, Bw[2]); v8f i3 = WMMA(Ax, Bw[3]);
    v8f i4 = WMMA(Ax, Bw[4]); v8f i5 = WMMA(Ax, Bw[5]);
    v8f g0 = WMMA(Ah, Bh[0]); v8f g1 = WMMA(Ah, Bh[1]);
    v8f g2 = WMMA(Ah, Bh[2]); v8f g3 = WMMA(Ah, Bh[3]);
    v8f g4 = WMMA(Ah, Bh[4]); v8f g5 = WMMA(Ah, Bh[5]);
    const float rA = fsig (i0[0] + dbi[0][0] + g0[0] + dbh[0][0]);
    const float zA = fsig (i2[0] + dbi[1][0] + g2[0] + dbh[1][0]);
    const float nA = ftanh_(i4[0] + dbi[2][0] + rA * (g4[0] + dbh[2][0]));
    const float newA = (1.0f - zA) * nA + zA * hhA;
    const float rB = fsig (i1[0] + dbi[0][1] + g1[0] + dbh[0][1]);
    const float zB = fsig (i3[0] + dbi[1][1] + g3[0] + dbh[1][1]);
    const float nB = ftanh_(i5[0] + dbi[2][1] + rB * (g5[0] + dbh[2][1]));
    const float newB = (1.0f - zB) * nB + zB * hhB;
    hxA = hhA = newA; hxB = hhB = newB;

    // res[s] = lin_W . new_h + lin_b  (lanes 16-31 mirror -> mask them to 0)
    float p = (lane < 16) ? (lwA * newA + lwB * newB) : 0.0f;
    p += __shfl_xor(p, 16, 32);
    p += __shfl_xor(p,  8, 32);
    p += __shfl_xor(p,  4, 32);
    p += __shfl_xor(p,  2, 32);
    p += __shfl_xor(p,  1, 32);
    if (lane == 0) out[s] = p + lb;
  }
}

extern "C" void kernel_launch(void* const* d_in, const int* in_sizes, int n_in,
                              void* d_out, int out_size, void* d_ws, size_t ws_size,
                              hipStream_t stream) {
  const float* x    = (const float*)d_in[0];
  const float* eWih = (const float*)d_in[1];
  const float* eWhh = (const float*)d_in[2];
  const float* ebih = (const float*)d_in[3];
  const float* ebhh = (const float*)d_in[4];
  const float* dWih = (const float*)d_in[5];
  const float* dWhh = (const float*)d_in[6];
  const float* dbih = (const float*)d_in[7];
  const float* dbhh = (const float*)d_in[8];
  const float* linW = (const float*)d_in[9];
  const float* linb = (const float*)d_in[10];
  const int n_enc = in_sizes[0];

  seq2seq_gru_kernel<<<dim3(1), dim3(32), 0, stream>>>(
      x, n_enc, eWih, eWhh, ebih, ebhh,
      dWih, dWhh, dbih, dbhh, linW, linb,
      (float*)d_out, out_size);
}
